// SelfAttentionInterview_7164005449951
// MI455X (gfx1250) — compile-verified, matrix operand order checked by reading, not run on hardware
//
#include <hip/hip_runtime.h>

typedef __bf16 bf16;
typedef __attribute__((ext_vector_type(16))) __bf16 v16bf;
typedef __attribute__((ext_vector_type(8)))  float  v8f;
typedef __attribute__((ext_vector_type(4)))  unsigned int u32x4;
typedef __attribute__((ext_vector_type(8)))  int  i32x8;
typedef __attribute__((ext_vector_type(4)))  int  i32x4;

#define D_DIM 256
#define N_SEQ 4096
#define B_SZ  4

// ---------------------------------------------------------------------------
// LDS layout (single allocation so TDM/async LDS byte offsets are known):
//   K double buffer : 2 x 32x256 bf16 = 32 KB   (filled by TDM)
//   V double buffer : 2 x 256x32 bf16 = 32 KB   (transposed [d][k])
//   Q tiles         : 8 waves x 16x256 bf16 = 64 KB (filled by async copy)
//   P staging       : 8 waves x 16x32 bf16  =  8 KB
//   total 136 KB (needs CDNA5's 320 KB WGP LDS)
// ---------------------------------------------------------------------------
#define KTILE_B   (32 * D_DIM * 2)
#define VTILE_B   (D_DIM * 32 * 2)
#define LDS_K0    0
#define LDS_K1    (LDS_K0 + KTILE_B)
#define LDS_V0    (LDS_K1 + KTILE_B)
#define LDS_V1    (LDS_V0 + VTILE_B)
#define LDS_Q     (LDS_V1 + VTILE_B)
#define QTILE_B   (16 * D_DIM * 2)
#define LDS_P     (LDS_Q + 8 * QTILE_B)
#define PTILE_B   (16 * 32 * 2)
#define LDS_TOTAL (LDS_P + 8 * PTILE_B)

// Async global->LDS copy, 16 bytes per lane (ASYNCcnt).
static __device__ __forceinline__ void async_copy_b128(unsigned lds_off, const void* gptr) {
  asm volatile("global_load_async_to_lds_b128 %0, %1, off"
               :: "v"(lds_off), "v"((unsigned long long)(uintptr_t)gptr)
               : "memory");
}
static __device__ __forceinline__ void wait_async0() {
  asm volatile("s_wait_asynccnt 0x0" ::: "memory");
}

// TDM: one descriptor-driven DMA of a 2D bf16 tile (rows x 256) global->LDS.
// D# bitfields per CDNA5 ISA.
// This toolchain exposes the 6-arg builtin: (g0, g1, g2, g3, i32x8, cpol).
static __device__ __forceinline__ void tdm_load_tile_rows256(
    unsigned lds_off, const void* gptr, int rows) {
  const unsigned long long ga = (unsigned long long)(uintptr_t)gptr;
  u32x4 g0 = {};
  i32x8 g1 = {};
  i32x4 g2 = {};
  i32x4 g3 = {};
  i32x8 g4 = {};
  g0[0] = 1u;                                         // count=1, load, user mode
  g0[1] = lds_off;                                    // lds_addr (bytes)
  g0[2] = (unsigned)(ga & 0xFFFFFFFFu);               // global_addr[31:0]
  g0[3] = (unsigned)((ga >> 32) & 0x1FFFFFFu)         // global_addr[56:32]
        | (2u << 30);                                 // type=2 ("image")
  g1[0] = (1 << 16);                                  // data_size=1 -> 2 bytes/elt
  g1[1] = (int)((D_DIM & 0xFFFF) << 16);              // tensor_dim0[15:0]
  g1[2] = (int)(((unsigned)D_DIM >> 16)               // tensor_dim0[31:16]
        | (((unsigned)rows & 0xFFFF) << 16));         // tensor_dim1[15:0]
  g1[3] = (int)(((unsigned)D_DIM & 0xFFFF) << 16);    // tensor_dim1[31:16]=0, tile_dim0
  g1[4] = rows;                                       // tile_dim1 (tile_dim2=0)
  g1[5] = D_DIM;                                      // tensor_dim0_stride[31:0]
  g1[6] = 0;                                          // stride hi, dim1_stride lo
  g1[7] = 0;
  __builtin_amdgcn_tensor_load_to_lds(g0, g1, g2, g3, g4, 0);
}

// ---------------------------------------------------------------------------
// Fragment loaders following the CDNA5 16-bit A/B WMMA VGPR layout:
// lane half h (lane>=16 -> h=1) has K-base = 8*h; vector elements
// j in [0,8)  hold K = kb + j        (contiguous 16B run)
// j in [8,16) hold K = kb + 16 + j-8 (contiguous 16B run)
// where kb = kchunk + 8*h. Caller passes p = row_base + kchunk + 8*h.
// ---------------------------------------------------------------------------
static __device__ __forceinline__ v16bf load_frag_f32(const float* __restrict__ p) {
  v16bf a;
#pragma unroll
  for (int j = 0; j < 8; ++j) {
    a[j]     = (bf16)p[j];
    a[j + 8] = (bf16)p[j + 16];
  }
  return a;
}

static __device__ __forceinline__ v16bf load_frag_bf16(const bf16* __restrict__ p) {
  v16bf a;
#pragma unroll
  for (int j = 0; j < 8; ++j) {
    a[j]     = p[j];
    a[j + 8] = p[j + 16];
  }
  return a;
}

static __device__ __forceinline__ float half16_max(float v) {
#pragma unroll
  for (int m = 1; m < 16; m <<= 1) v = fmaxf(v, __shfl_xor(v, m, 16));
  return v;
}
static __device__ __forceinline__ float half16_sum(float v) {
#pragma unroll
  for (int m = 1; m < 16; m <<= 1) v += __shfl_xor(v, m, 16);
  return v;
}

// ---------------------------------------------------------------------------
// Kernel 1: Q/K/V projections.  out[n,e] = sum_d x[n,d] * W[e,d] + b[e]
// B-operand column e == row e of W (row-major) -> contiguous loads, no transpose.
// One wave computes a 16(M) x 32(N) slab; grid.x = BN/16, grid.y = 3.
// ---------------------------------------------------------------------------
__global__ __launch_bounds__(256) void qkv_proj_kernel(
    const float* __restrict__ x,
    const float* __restrict__ Wq, const float* __restrict__ bq,
    const float* __restrict__ Wk, const float* __restrict__ bk,
    const float* __restrict__ Wv, const float* __restrict__ bv,
    bf16* __restrict__ Q, bf16* __restrict__ K, bf16* __restrict__ V) {
  const int lane = threadIdx.x & 31;
  const int wave = threadIdx.x >> 5;
  const int hh   = lane >> 4;     // lane half
  const int lid  = lane & 15;
  const int m0   = blockIdx.x * 16;
  const int mat  = blockIdx.y;

  const float* W    = (mat == 0) ? Wq : (mat == 1) ? Wk : Wv;
  const float* bias = (mat == 0) ? bq : (mat == 1) ? bk : bv;
  bf16*        out  = (mat == 0) ? Q  : (mat == 1) ? K  : V;

  const int arow = m0 + lid;
  const int n0 = (wave * 2 + 0) * 16;
  const int n1 = (wave * 2 + 1) * 16;

  v8f acc0 = {};
  v8f acc1 = {};
#pragma unroll
  for (int k0 = 0; k0 < D_DIM; k0 += 32) {
    const int kb = k0 + hh * 8;
    v16bf a  = load_frag_f32(x + (size_t)arow * D_DIM + kb);
    v16bf b0 = load_frag_f32(W + (size_t)(n0 + lid) * D_DIM + kb);
    v16bf b1 = load_frag_f32(W + (size_t)(n1 + lid) * D_DIM + kb);
    acc0 = __builtin_amdgcn_wmma_f32_16x16x32_bf16(false, a, false, b0, (short)0, acc0, false, false);
    acc1 = __builtin_amdgcn_wmma_f32_16x16x32_bf16(false, a, false, b1, (short)0, acc1, false, false);
  }

  const float bv0 = bias[n0 + lid];
  const float bv1 = bias[n1 + lid];
#pragma unroll
  for (int r = 0; r < 8; ++r) {
    const int row = m0 + r + hh * 8;   // C layout: VGPR r, lane-half h -> M = r + 8h
    out[(size_t)row * D_DIM + n0 + lid] = (bf16)(acc0[r] + bv0);
    out[(size_t)row * D_DIM + n1 + lid] = (bf16)(acc1[r] + bv1);
  }
}

// ---------------------------------------------------------------------------
// Kernel 2: flash attention. Block = 8 waves = 128 queries; wave owns 16
// queries with full 16x256 f32 accumulator in VGPRs. K tiles staged by the
// Tensor Data Mover (one DMA per tile, TENSORcnt), Q tiles by async
// global->LDS (ASYNCcnt), V tiles transposed on store; K/V double-buffered
// so staging of tile kt+1 overlaps compute of tile kt.
// ---------------------------------------------------------------------------
__global__ __launch_bounds__(256) void flash_attn_kernel(
    const bf16* __restrict__ Q, const bf16* __restrict__ K,
    const bf16* __restrict__ V, const int* __restrict__ mask,
    float* __restrict__ out) {
  __shared__ __align__(16) char smem[LDS_TOTAL];

  const int tid  = threadIdx.x;
  const int lane = tid & 31;
  const int wave = tid >> 5;
  const int hh   = lane >> 4;
  const int lid  = lane & 15;
  const int b    = blockIdx.y;
  const int q0   = blockIdx.x * 128 + wave * 16;

  const bf16* Qb = Q + (size_t)b * N_SEQ * D_DIM;
  const bf16* Kb = K + (size_t)b * N_SEQ * D_DIM;
  const bf16* Vb = V + (size_t)b * N_SEQ * D_DIM;

  const bf16* Qs = (const bf16*)(smem + LDS_Q + wave * QTILE_B);
  bf16*       Pw = (bf16*)(smem + LDS_P + wave * PTILE_B);

  // ---- Stage this wave's 16 query rows via async copy (512 x 16B) ----
  {
    const char* src = (const char*)(Qb + (size_t)q0 * D_DIM);
    const unsigned dst = LDS_Q + wave * QTILE_B;
    for (int i = lane; i < 512; i += 32)
      async_copy_b128(dst + i * 16, src + i * 16);
  }

  // ---- K tile: single TDM DMA per tile, issued by wave 0 only ----
  auto stage_k = [&](int kt, int buf) {
    if (wave == 0)
      tdm_load_tile_rows256(buf ? LDS_K1 : LDS_K0, Kb + (size_t)kt * 32 * D_DIM, 32);
  };
  // ---- V tile: VGPR-path transpose into [d][k] layout ----
  auto stage_v = [&](int kt, int buf) {
    bf16* Vt = (bf16*)(smem + (buf ? LDS_V1 : LDS_V0));
    for (int c = tid; c < 1024; c += 256) {          // 1024 x 8 elements
      const int kk  = c >> 5;
      const int off = (c & 31) * 8;
      const bf16* vs = Vb + (size_t)(kt * 32 + kk) * D_DIM + off;
#pragma unroll
      for (int j = 0; j < 8; ++j) Vt[(off + j) * 32 + kk] = vs[j];  // transpose
    }
  };

  float mrun[8], lsum[8];
#pragma unroll
  for (int r = 0; r < 8; ++r) { mrun[r] = -__builtin_inff(); lsum[r] = 0.0f; }

  const v8f czero = {};
  v8f o[16];
#pragma unroll
  for (int ct = 0; ct < 16; ++ct) o[ct] = czero;

  const float scale = 0.0625f;  // 1/sqrt(256)
  const int NT = N_SEQ / 32;

  stage_k(0, 0);
  stage_v(0, 0);
  if (wave == 0) __builtin_amdgcn_s_wait_tensorcnt(0);  // K tile DMA done
  wait_async0();                                        // Q tile done (per wave)
  __syncthreads();                                      // all staging visible

  for (int kt = 0; kt < NT; ++kt) {
    const int buf = kt & 1;
    if (kt + 1 < NT) {                      // overlap next-tile staging
      stage_k(kt + 1, buf ^ 1);
      stage_v(kt + 1, buf ^ 1);
    }

    const bf16* Kt = (const bf16*)(smem + (buf ? LDS_K1 : LDS_K0));
    const bf16* Vt = (const bf16*)(smem + (buf ? LDS_V1 : LDS_V0));

    // ---- S = Q * Ktile^T : 16 queries x 32 keys as two 16x16 C tiles ----
    v8f s0 = {};
    v8f s1 = {};
#pragma unroll
    for (int k0 = 0; k0 < D_DIM; k0 += 32) {
      const int kb = k0 + hh * 8;
      v16bf a  = load_frag_bf16(&Qs[lid * D_DIM + kb]);
      v16bf b0 = load_frag_bf16(&Kt[(lid)      * D_DIM + kb]);  // keys 0..15
      v16bf b1 = load_frag_bf16(&Kt[(16 + lid) * D_DIM + kb]);  // keys 16..31
      s0 = __builtin_amdgcn_wmma_f32_16x16x32_bf16(false, a, false, b0, (short)0, s0, false, false);
      s1 = __builtin_amdgcn_wmma_f32_16x16x32_bf16(false, a, false, b1, (short)0, s1, false, false);
    }

    // ---- online softmax (per query row, rows live in 16-lane halves) ----
    float alpha[8];
#pragma unroll
    for (int r = 0; r < 8; ++r) {
      const int qrow = q0 + r + hh * 8;
      const int* mrow = mask + ((size_t)b * N_SEQ + qrow) * N_SEQ + (size_t)kt * 32;
      float sv0 = (mrow[lid]      == 0) ? -__builtin_inff() : s0[r] * scale;
      float sv1 = (mrow[16 + lid] == 0) ? -__builtin_inff() : s1[r] * scale;
      float tmax = half16_max(fmaxf(sv0, sv1));
      float mn = fmaxf(mrun[r], tmax);
      float al = __expf(mrun[r] - mn);
      float p0 = __expf(sv0 - mn);
      float p1 = __expf(sv1 - mn);
      float rs = half16_sum(p0 + p1);
      lsum[r] = lsum[r] * al + rs;
      mrun[r] = mn;
      alpha[r] = al;
      Pw[(r + hh * 8) * 32 + lid]      = (bf16)p0;  // C-layout -> row-major
      Pw[(r + hh * 8) * 32 + 16 + lid] = (bf16)p1;
    }

    // In-wave LDS RAW on Pw: CDNA5 split DS counter wait.
    asm volatile("s_wait_dscnt 0x0" ::: "memory");

    // ---- O = diag(alpha)*O + P * Vtile ----
    v16bf pf = load_frag_bf16(&Pw[lid * 32 + hh * 8]);  // A operand 16x32
#pragma unroll
    for (int ct = 0; ct < 16; ++ct) {
#pragma unroll
      for (int r = 0; r < 8; ++r) o[ct][r] *= alpha[r];
      v16bf vf = load_frag_bf16(&Vt[(ct * 16 + lid) * 32 + hh * 8]);  // B: col d, K=key
      o[ct] = __builtin_amdgcn_wmma_f32_16x16x32_bf16(false, pf, false, vf, (short)0, o[ct], false, false);
    }

    // Wave 0 drains the next K tile's TDM; V-path ds stores drain at barrier.
    if (wave == 0) __builtin_amdgcn_s_wait_tensorcnt(0);
    __syncthreads();
  }

  // ---- normalize and write f32 output ----
  float* ob = out + (size_t)b * N_SEQ * D_DIM;
#pragma unroll
  for (int r = 0; r < 8; ++r) {
    const int qrow = q0 + r + hh * 8;
    const float inv = 1.0f / lsum[r];
#pragma unroll
    for (int ct = 0; ct < 16; ++ct) {
      ob[(size_t)qrow * D_DIM + ct * 16 + lid] = o[ct][r] * inv;
    }
  }
}

// ---------------------------------------------------------------------------
extern "C" void kernel_launch(void* const* d_in, const int* in_sizes, int n_in,
                              void* d_out, int out_size, void* d_ws, size_t ws_size,
                              hipStream_t stream) {
  (void)in_sizes; (void)n_in; (void)out_size; (void)ws_size;
  const float* x   = (const float*)d_in[0];
  const int* mask  = (const int*)d_in[1];
  const float* Wq  = (const float*)d_in[2];
  const float* bq  = (const float*)d_in[3];
  const float* Wk  = (const float*)d_in[4];
  const float* bk  = (const float*)d_in[5];
  const float* Wv  = (const float*)d_in[6];
  const float* bv  = (const float*)d_in[7];
  float* out = (float*)d_out;

  // Workspace: Q, K, V in bf16, 8 MB each = 24 MB.
  bf16* Qw = (bf16*)d_ws;
  bf16* Kw = Qw + (size_t)B_SZ * N_SEQ * D_DIM;
  bf16* Vw = Kw + (size_t)B_SZ * N_SEQ * D_DIM;

  dim3 gProj(B_SZ * N_SEQ / 16, 3, 1);
  qkv_proj_kernel<<<gProj, 256, 0, stream>>>(x, Wq, bq, Wk, bk, Wv, bv, Qw, Kw, Vw);

  dim3 gAttn(N_SEQ / 128, B_SZ, 1);
  flash_attn_kernel<<<gAttn, 256, 0, stream>>>(Qw, Kw, Vw, mask, out);
}
